// TriangleMultiplication_5531917877747
// MI455X (gfx1250) — compile-verified, matrix operand order checked by reading, not run on hardware
//
#include <hip/hip_runtime.h>
#include <hip/hip_bf16.h>
#include <math.h>

// ---------------------------------------------------------------------------
// TriangleMultiplication (outgoing) for MI455X / gfx1250, wave32 + WMMA bf16.
//   z:(1,512,512,144) f32 -> out:(1,512,512,144) f32
// Pipeline: k0 weight pack (bf16) -> k1 fused LN+projection GEMM (WMMA)
//           -> k2 per-channel triangle GEMM (WMMA) -> k3 LN+output GEMM (WMMA)
// Intermediates a_t/b_t are bf16 channel-major (67MB each: both L2-resident).
// ---------------------------------------------------------------------------

typedef __bf16 bf16;
typedef __attribute__((ext_vector_type(16))) __bf16 v16bf;
typedef __attribute__((ext_vector_type(8)))  float  v8f;

#define LL 512          // sequence length
#define DIMZ 144        // pair dim
#define CH 128          // hidden channels
#define NCAT 656        // 256 (W_ab) + 256 (W_abg) + 144 (W_g)
#define KPAD 160        // 144 padded to 5 * 32

__device__ __forceinline__ v8f wmma_bf16(v16bf a, v16bf b, v8f c) {
  // D = A(16x32 bf16) * B(32x16 bf16) + C(16x16 f32)
  return __builtin_amdgcn_wmma_f32_16x16x32_bf16(
      false, a, false, b, (short)0, c, false, false);
}

// Fast sigmoid: v_exp_f32 + v_rcp_f32 (result feeds bf16-precision data, so
// the ~1ulp rcp is plenty; avoids the IEEE div_scale/div_fmas chain).
__device__ __forceinline__ float sigmoidf_(float x) {
  return __builtin_amdgcn_rcpf(1.0f + __expf(-x));
}

// ---------------------------------------------------------------------------
// k0: pack weights into bf16.
//  wcat[n][d] (NCAT x KPAD): n<256 -> W_ab[n], n<512 -> W_abg[n-256],
//                            else W_g[n-512]; d>=144 zero-padded.
//  woutb[m][c] (144 x 128)
// ---------------------------------------------------------------------------
__global__ void k0_prep(const float* __restrict__ Wab,
                        const float* __restrict__ Wabg,
                        const float* __restrict__ Wg,
                        const float* __restrict__ Wout,
                        bf16* __restrict__ wcat, bf16* __restrict__ woutb) {
  int stride = gridDim.x * blockDim.x;
  int tid = blockIdx.x * blockDim.x + threadIdx.x;
  for (int idx = tid; idx < NCAT * KPAD; idx += stride) {
    int n = idx / KPAD, d = idx % KPAD;
    float v = 0.0f;
    if (d < DIMZ) {
      if (n < 256)      v = Wab[n * DIMZ + d];
      else if (n < 512) v = Wabg[(n - 256) * DIMZ + d];
      else              v = Wg[(n - 512) * DIMZ + d];
    }
    wcat[idx] = (bf16)v;
  }
  for (int idx = tid; idx < DIMZ * CH; idx += stride)
    woutb[idx] = (bf16)Wout[idx];
}

// ---------------------------------------------------------------------------
// k1: fused LayerNorm(z) + projection GEMM.
//  Block = 64 flattened (i,k) rows. A = wcat rows (M=out-channel),
//  B = zn rows (N=row) so D lanes index rows -> k-contiguous stores into
//  a_t[c][i][k] / b_t[c][j][k] (bf16, channel-major for k2).
// ---------------------------------------------------------------------------
__global__ __launch_bounds__(256) void k1_proj(
    const float* __restrict__ z, const float* __restrict__ pmask,
    const float* __restrict__ ng, const float* __restrict__ nb,
    const bf16* __restrict__ wcat,
    bf16* __restrict__ at, bf16* __restrict__ bt, float* __restrict__ g) {
  __shared__ alignas(16) float raw[64 * 148];   // 37.9 KB
  __shared__ alignas(32) bf16  zn[64 * KPAD];   // 20.5 KB
  int t = threadIdx.x;
  long rowBase = (long)blockIdx.x * 64;

  // Phase A: coalesced float4 load of 64 rows x 144 into padded LDS.
  const float4* zsrc = (const float4*)(z + rowBase * DIMZ);
  for (int i4 = t; i4 < 64 * 36; i4 += 256) {
    int r = i4 / 36, c4 = i4 % 36;
    *(float4*)&raw[r * 148 + c4 * 4] = zsrc[i4];
  }
  __syncthreads();

  // Phase B: per-row LayerNorm -> bf16 tile (K zero-padded to 160).
  if (t < 64) {
    const float* row = &raw[t * 148];
    float s = 0.0f;
    for (int c = 0; c < DIMZ; ++c) s += row[c];
    float mu = s * (1.0f / DIMZ);
    float v = 0.0f;
    for (int c = 0; c < DIMZ; ++c) { float d = row[c] - mu; v += d * d; }
    float inv = rsqrtf(v * (1.0f / DIMZ) + 1e-5f);
    bf16* dst = &zn[t * KPAD];
    for (int c = 0; c < DIMZ; ++c)
      dst[c] = (bf16)((row[c] - mu) * inv * ng[c] + nb[c]);
    for (int c = DIMZ; c < KPAD; ++c) dst[c] = (bf16)0.0f;
  }
  __syncthreads();

  // Phase C: WMMA. Units: 64 value+gate pairs (16 m-pairs x 4 n-tiles)
  //                       + 36 g units (9 m-tiles x 4 n-tiles).
  int wave = t >> 5, lane = t & 31, ln = lane & 15, hi = lane >> 4;
  for (int u = wave; u < 100; u += 8) {            // wave-uniform loop
    int nt = (u < 64) ? (u & 3) : ((u - 64) & 3);
    const bf16* bsrc = &zn[(nt * 16 + ln) * KPAD + hi * 16];
    long rg = rowBase + nt * 16 + ln;              // this lane's (i,k) row
    float mk = pmask[rg];
    long ii = rg >> 9, kk0 = rg & 511;
    if (u < 64) {
      int mp = u >> 2;                             // channel pair tile 0..15
      const bf16* av = &wcat[(mp * 16 + ln) * KPAD + hi * 16];
      const bf16* ag = &wcat[(256 + mp * 16 + ln) * KPAD + hi * 16];
      v8f accV = {}; v8f accG = {};
      for (int ks = 0; ks < 5; ++ks) {
        v16bf B  = *(const v16bf*)(bsrc + ks * 32);
        v16bf A1 = *(const v16bf*)(av + ks * 32);
        v16bf A2 = *(const v16bf*)(ag + ks * 32);
        accV = wmma_bf16(A1, B, accV);
        accG = wmma_bf16(A2, B, accG);
      }
      for (int r = 0; r < 8; ++r) {
        int ch = mp * 16 + 8 * hi + r;             // output channel 0..255
        float val = accV[r] * sigmoidf_(accG[r]) * mk;
        bf16* dst = (ch < CH) ? at : bt;
        long c2 = ch & (CH - 1);
        dst[(c2 << 18) + (ii << 9) + kk0] = (bf16)val;
      }
    } else {
      int mg = (u - 64) >> 2;                      // g tile 0..8
      const bf16* aw = &wcat[(512 + mg * 16 + ln) * KPAD + hi * 16];
      v8f acc = {};
      for (int ks = 0; ks < 5; ++ks) {
        v16bf B = *(const v16bf*)(bsrc + ks * 32);
        v16bf A = *(const v16bf*)(aw + ks * 32);
        acc = wmma_bf16(A, B, acc);
      }
      float* gd = &g[rg * DIMZ + mg * 16 + 8 * hi];
      for (int r = 0; r < 8; ++r) gd[r] = sigmoidf_(acc[r]);
    }
  }
}

// ---------------------------------------------------------------------------
// k2: triangle einsum out[i,j,c] = sum_k a[i,k,c]*b[j,k,c] per channel.
//  Block: 32i x 32j x 32c; 8 waves x 4 channels; 2x2 WMMA tiles, K=512.
//  a_t/b_t (134MB total bf16) are L2-resident (192MB). Output [c][i][j],
//  j contiguous across lanes -> coalesced f32 stores. Mask fused pre-LN.
// ---------------------------------------------------------------------------
__global__ __launch_bounds__(256) void k2_tri(
    const bf16* __restrict__ at, const bf16* __restrict__ bt,
    const float* __restrict__ pmask, float* __restrict__ out2) {
  __shared__ float mt[32 * 33];
  int t = threadIdx.x;
  int i0 = blockIdx.x * 32, j0 = blockIdx.y * 32, cB = blockIdx.z * 32;
  for (int p = t; p < 32 * 32; p += 256) {
    int r = p >> 5, cc = p & 31;
    mt[r * 33 + cc] = pmask[(long)(i0 + r) * LL + j0 + cc];
  }
  __syncthreads();
  int wave = t >> 5, lane = t & 31, ln = lane & 15, hi = lane >> 4;
  for (int cc = 0; cc < 4; ++cc) {                 // wave-uniform
    long c = cB + wave * 4 + cc;
    const bf16* ab = at + (c << 18) + ((long)(i0 + ln) << 9) + (hi << 4);
    const bf16* bb = bt + (c << 18) + ((long)(j0 + ln) << 9) + (hi << 4);
    v8f a00 = {}, a01 = {}, a10 = {}, a11 = {};
    for (int ks = 0; ks < 16; ++ks) {
      int kb = ks << 5;
      v16bf A0 = *(const v16bf*)(ab + kb);
      v16bf A1 = *(const v16bf*)(ab + (16 << 9) + kb);
      v16bf B0 = *(const v16bf*)(bb + kb);
      v16bf B1 = *(const v16bf*)(bb + (16 << 9) + kb);
      a00 = wmma_bf16(A0, B0, a00);
      a01 = wmma_bf16(A0, B1, a01);
      a10 = wmma_bf16(A1, B0, a10);
      a11 = wmma_bf16(A1, B1, a11);
    }
    float* ob = out2 + (c << 18);
    for (int ti = 0; ti < 2; ++ti)
      for (int tj = 0; tj < 2; ++tj) {
        v8f acc = (ti == 0) ? (tj == 0 ? a00 : a01) : (tj == 0 ? a10 : a11);
        int jl = tj * 16 + ln;
        for (int r = 0; r < 8; ++r) {
          int il = ti * 16 + 8 * hi + r;
          ob[(long)(i0 + il) * LL + j0 + jl] = acc[r] * mt[il * 33 + jl];
        }
      }
  }
}

// ---------------------------------------------------------------------------
// k3: LayerNorm over C + output GEMM (A = W_out, M=144, K=128, N=64 pos),
//  fused *g*mask, LDS transpose -> fully coalesced 151MB f32 store.
//  reg0 is reused: raw [64][133] f32 tile, then out [64][148] f32 tile.
// ---------------------------------------------------------------------------
__global__ __launch_bounds__(256) void k3_out(
    const float* __restrict__ out2, const float* __restrict__ g,
    const float* __restrict__ pmask, const float* __restrict__ n1g,
    const float* __restrict__ n1b, const bf16* __restrict__ woutb,
    float* __restrict__ out) {
  __shared__ alignas(16) float reg0[64 * 148];    // 37.9 KB (raw -> out)
  __shared__ alignas(32) bf16  lnT[64 * DIMZ];    // 18.4 KB (stride 144)
  int t = threadIdx.x;
  long p0 = (long)blockIdx.x * 64;

  // Transposing load: out2 is [c][p]; 64 consecutive p per c are contiguous.
  {
    int cSub = t >> 6, pos = t & 63;
    for (int cb = 0; cb < CH; cb += 4) {
      long c = cb + cSub;
      reg0[pos * 133 + (int)c] = out2[(c << 18) + p0 + pos];
    }
  }
  __syncthreads();

  if (t < 64) {
    const float* row = &reg0[t * 133];
    float s = 0.0f;
    for (int c = 0; c < CH; ++c) s += row[c];
    float mu = s * (1.0f / CH);
    float v = 0.0f;
    for (int c = 0; c < CH; ++c) { float d = row[c] - mu; v += d * d; }
    float inv = rsqrtf(v * (1.0f / CH) + 1e-5f);
    bf16* dst = &lnT[t * DIMZ];
    for (int c = 0; c < CH; ++c)
      dst[c] = (bf16)((row[c] - mu) * inv * n1g[c] + n1b[c]);
  }
  __syncthreads();

  int wave = t >> 5, lane = t & 31, ln = lane & 15, hi = lane >> 4;
  for (int u = wave; u < 36; u += 8) {             // 9 m-tiles x 4 n-tiles
    int mtile = u / 4, nt = u & 3, m0 = mtile * 16;
    const bf16* bsrc = &lnT[(nt * 16 + ln) * DIMZ + hi * 16];
    const bf16* asrc = &woutb[(m0 + ln) * CH + hi * 16];
    v8f acc = {};
    for (int ks = 0; ks < 4; ++ks) {
      v16bf A = *(const v16bf*)(asrc + ks * 32);
      v16bf B = *(const v16bf*)(bsrc + ks * 32);
      acc = wmma_bf16(A, B, acc);
    }
    int pl = nt * 16 + ln;
    long pg = p0 + pl;
    float mk = pmask[pg];
    const float* gv = &g[pg * DIMZ + m0 + 8 * hi];
    float* od = &reg0[pl * 148 + m0 + 8 * hi];
    for (int r = 0; r < 8; ++r) od[r] = acc[r] * gv[r] * mk;
  }
  __syncthreads();

  for (int f = t; f < 64 * DIMZ; f += 256) {       // coalesced final store
    int pos = f / DIMZ, dim = f - pos * DIMZ;
    out[p0 * DIMZ + f] = reg0[pos * 148 + dim];
  }
}

// ---------------------------------------------------------------------------
// Workspace layout (bytes):
//   wcat  bf16 656*160           @ 0          (209,920)
//   woutb bf16 144*128           @ 209,920    ( 36,864)
//   a_t   bf16 128*512*512       @ 246,784    (67,108,864)
//   b_t   bf16 128*512*512       @ 67,355,648 (67,108,864)
//   g     f32  262144*144        @ 134,464,512 (150,994,944)
//   out2  f32  128*512*512       @ 285,459,456 (134,217,728)  total ~400 MB
// ---------------------------------------------------------------------------
extern "C" void kernel_launch(void* const* d_in, const int* in_sizes, int n_in,
                              void* d_out, int out_size, void* d_ws,
                              size_t ws_size, hipStream_t stream) {
  (void)in_sizes; (void)n_in; (void)out_size; (void)ws_size;
  const float* z    = (const float*)d_in[0];
  const float* pm   = (const float*)d_in[1];
  const float* ng   = (const float*)d_in[2];
  const float* nb   = (const float*)d_in[3];
  const float* n1g  = (const float*)d_in[4];
  const float* n1b  = (const float*)d_in[5];
  const float* Wab  = (const float*)d_in[6];
  const float* Wabg = (const float*)d_in[7];
  const float* Wg   = (const float*)d_in[8];
  const float* Wout = (const float*)d_in[9];
  float* out = (float*)d_out;

  char* ws = (char*)d_ws;
  bf16*  wcat  = (bf16*)(ws + 0);
  bf16*  woutb = (bf16*)(ws + 209920);
  bf16*  a_t   = (bf16*)(ws + 246784);
  bf16*  b_t   = (bf16*)(ws + 67355648);
  float* gbuf  = (float*)(ws + 134464512);
  float* out2  = (float*)(ws + 285459456);

  k0_prep<<<128, 256, 0, stream>>>(Wab, Wabg, Wg, Wout, wcat, woutb);
  k1_proj<<<4096, 256, 0, stream>>>(z, pm, ng, nb, wcat, a_t, b_t, gbuf);
  k2_tri<<<dim3(16, 16, 4), 256, 0, stream>>>(a_t, b_t, pm, out2);
  k3_out<<<4096, 256, 0, stream>>>(out2, gbuf, pm, n1g, n1b, woutb, out);
}
